// CaptionGenerationModule_34419867910855
// MI455X (gfx1250) — matrix operand order repaired and checked by reading, not compiled, not run on hardware
//
#include <hip/hip_runtime.h>
#include <math.h>

// ---------------------------------------------------------------------------
// Problem constants
// ---------------------------------------------------------------------------
#define V_     16000
#define MAXLEN 28
#define TDEC   27          // MAXLEN-1 decode steps
#define DW_    512
#define DV_    2048
#define DH_    512
#define B_     128
#define NF_    40
#define G3     1536        // 3*DH

// ---------------------------------------------------------------------------
// WMMA bf16 helpers (CDNA5 gfx1250, wave32)
// ---------------------------------------------------------------------------
typedef __attribute__((ext_vector_type(16))) __bf16 v16bf;
typedef __attribute__((ext_vector_type(8)))  float  v8f;

union FragBF { uint4 q[2]; v16bf v; };

// A fragment: 16x32 bf16 tile, row-major src, ld in elements.
// lane (0..15): row=lane, elems 0..7 = K[0..7],  8..15 = K[16..23]
// lane(16..31): row=lane-16, elems 0..7 = K[8..15], 8..15 = K[24..31]
// caller passes p = base + row*ld + koff   (koff = (lane>>4)*8)
__device__ __forceinline__ v16bf ld_afrag(const __bf16* p) {
    FragBF f;
    f.q[0] = *(const uint4*)(p);
    f.q[1] = *(const uint4*)(p + 16);
    return f.v;
}
// B fragment: 32x16 bf16 tile taken from row-major W[N,K] (B[k][n] = W[n][k]).
// lane(0..15): col n=lane, elems 0..7 = K[0..7],  8..15 = K[16..23]
// lane(16..31): col n=lane-16, elems 0..7 = K[8..15], 8..15 = K[24..31]
// caller passes p = W + (n0+n)*ldw + k0 + koff (koff = (lane>>4)*8)
__device__ __forceinline__ v16bf ld_bfrag(const __bf16* p) {
    FragBF f;
    f.q[0] = *(const uint4*)(p);
    f.q[1] = *(const uint4*)(p + 16);
    return f.v;
}
__device__ __forceinline__ v8f wmma_bf16(v16bf a, v16bf b, v8f c) {
    return __builtin_amdgcn_wmma_f32_16x16x32_bf16(false, a, false, b,
                                                   (short)0, c, false, false);
}
__device__ __forceinline__ float sigmoidf_(float x) {
    return 1.0f / (1.0f + __expf(-x));
}
#define ZERO8 {0.f,0.f,0.f,0.f,0.f,0.f,0.f,0.f}

// ---------------------------------------------------------------------------
// fp32 -> bf16 convert
// ---------------------------------------------------------------------------
__global__ void f2bf_kernel(const float* __restrict__ s, __bf16* __restrict__ d, int n) {
    int i = blockIdx.x * 256 + threadIdx.x;
    if (i < n) d[i] = (__bf16)s[i];
}

// vid_feats [B,NF,DV] -> Xenc_bf row (t*128+b) = vid[b,t,:]
__global__ void permute_vid_kernel(const float* __restrict__ vid, __bf16* __restrict__ X) {
    int idx = blockIdx.x * 256 + threadIdx.x;
    if (idx >= NF_ * B_ * DV_) return;
    int k   = idx & (DV_ - 1);
    int row = idx >> 11;          // DV=2048
    int t   = row >> 7;           // 128 rows per step
    int b   = row & 127;
    X[idx] = (__bf16)vid[((size_t)b * NF_ + t) * DV_ + k];
}

// word gather: Wd row (t*128+b) = emb[target[b, t]] (t < 27), bf16
__global__ void gather_words_kernel(const int* __restrict__ tgt,
                                    const float* __restrict__ emb,
                                    __bf16* __restrict__ Wd) {
    int idx = blockIdx.x * 256 + threadIdx.x;
    if (idx >= TDEC * B_ * DW_) return;
    int k   = idx & (DW_ - 1);
    int row = idx >> 9;           // DW=512
    int t   = row >> 7;
    int b   = row & 127;
    int wid = tgt[b * MAXLEN + t];
    Wd[idx] = (__bf16)emb[(size_t)wid * DW_ + k];
}

// ---------------------------------------------------------------------------
// Generic bf16 WMMA GEMM:  C[M,N] = A[M,K] * W[N,K]^T + bias[N]
// block = 256 threads (8 waves); block tile 64(M) x 128(N); wave tile 32x32.
// Grid: (N/128, M/64). K % 32 == 0.
// ---------------------------------------------------------------------------
__global__ __launch_bounds__(256) void gemm_bf16_wmma(
    const __bf16* __restrict__ A, int lda,
    const __bf16* __restrict__ W, int ldw,
    const float*  __restrict__ bias,
    float* __restrict__ C, int ldc, int K)
{
    const int lane = threadIdx.x & 31;
    const int wave = threadIdx.x >> 5;
    const int wm = wave >> 2, wn = wave & 3;
    const int m0 = blockIdx.y * 64 + wm * 32;
    const int n0 = blockIdx.x * 128 + wn * 32;

    const int arow  = lane & 15;
    const int akoff = (lane >> 4) << 3;
    const int bkoff = (lane >> 4) << 3;

    const __bf16* A0 = A + (size_t)(m0 + arow) * lda + akoff;
    const __bf16* A1 = A0 + (size_t)16 * lda;
    const __bf16* W0 = W + (size_t)(n0 + arow) * ldw + bkoff;
    const __bf16* W1 = W0 + (size_t)16 * ldw;

    v8f c00 = ZERO8, c01 = ZERO8, c10 = ZERO8, c11 = ZERO8;

    for (int k0 = 0; k0 < K; k0 += 32) {
        v16bf a0 = ld_afrag(A0 + k0);
        v16bf a1 = ld_afrag(A1 + k0);
        v16bf b0 = ld_bfrag(W0 + k0);
        v16bf b1 = ld_bfrag(W1 + k0);
        c00 = wmma_bf16(a0, b0, c00);
        c01 = wmma_bf16(a0, b1, c01);
        c10 = wmma_bf16(a1, b0, c10);
        c11 = wmma_bf16(a1, b1, c11);
    }

    const int mrow = m0 + ((lane >> 4) << 3);
    const int ncol = n0 + (lane & 15);
    const float bz0 = bias[ncol];
    const float bz1 = bias[ncol + 16];
#pragma unroll
    for (int r = 0; r < 8; ++r) {
        C[(size_t)(mrow + r)      * ldc + ncol]      = c00[r] + bz0;
        C[(size_t)(mrow + r)      * ldc + ncol + 16] = c01[r] + bz1;
        C[(size_t)(mrow + 16 + r) * ldc + ncol]      = c10[r] + bz0;
        C[(size_t)(mrow + 16 + r) * ldc + ncol + 16] = c11[r] + bz1;
    }
}

// ---------------------------------------------------------------------------
// Recurrent kernel: one workgroup, 32 waves, h1/h2 state in 256KB LDS (bf16).
// For each of 67 steps: 3 gate-triple GEMMs [128,1536]x[512] + GRU elementwise.
// Wave w owns gate column tile jt = w (16 cols), loops over all 8 M-tiles.
// ---------------------------------------------------------------------------
__global__ __launch_bounds__(1024) void recurrent_kernel(
    const __bf16* __restrict__ Whh1,   // [1536,512]
    const __bf16* __restrict__ Wih2,   // [1536,1024] (cols 0..511 = h1 part)
    const __bf16* __restrict__ Whh2,   // [1536,512]
    const float*  __restrict__ Genc,   // [40*128, 1536] (incl. b_ih1)
    const float*  __restrict__ Gword,  // [27*128, 1536] (incl. b_ih2)
    const float*  __restrict__ b_ih1,
    const float*  __restrict__ b_hh1,
    const float*  __restrict__ b_ih2,
    const float*  __restrict__ b_hh2,
    __bf16* __restrict__ H2dec)        // [B*27, 512], row = b*27 + t
{
    __shared__ __bf16 h1s[B_ * DH_];   // 128 KB
    __shared__ __bf16 h2s[B_ * DH_];   // 128 KB

    const int tid = threadIdx.x;
    for (int i = tid; i < B_ * DH_; i += 1024) {
        h1s[i] = (__bf16)0.0f;
        h2s[i] = (__bf16)0.0f;
    }
    __syncthreads();

    const int lane  = tid & 31;
    const int wave  = tid >> 5;       // 0..31 -> gate column tile
    const int jt    = wave;
    const int jcol  = jt * 16 + (lane & 15);
    const int arow  = lane & 15;
    const int akoff = (lane >> 4) << 3;
    const int bkoff = (lane >> 4) << 3;
    const int mbase = (lane >> 4) << 3;

    // Per-lane weight row pointers (gate g adds g*512*ld)
    const __bf16* whh1_p = Whh1 + (size_t)jcol * DH_ + bkoff;
    const __bf16* wih2_p = Wih2 + (size_t)jcol * (DH_ + DW_) + bkoff;
    const __bf16* whh2_p = Whh2 + (size_t)jcol * DH_ + bkoff;

    const float bh1_r = b_hh1[jcol], bh1_z = b_hh1[512 + jcol], bh1_n = b_hh1[1024 + jcol];
    const float bh2_r = b_hh2[jcol], bh2_z = b_hh2[512 + jcol], bh2_n = b_hh2[1024 + jcol];
    const float bi1_r = b_ih1[jcol], bi1_z = b_ih1[512 + jcol], bi1_n = b_ih1[1024 + jcol];
    const float bi2_r = b_ih2[jcol], bi2_z = b_ih2[512 + jcol], bi2_n = b_ih2[1024 + jcol];

#pragma unroll 1
    for (int t = 0; t < NF_ + TDEC; ++t) {
        const bool enc = (t < NF_);
        const int  td  = t - NF_;

        // ---------------- GRU1: gh1 = h1 @ Whh1^T ----------------
        __bf16 nh1[64];
#pragma unroll 1
        for (int p = 0; p < 8; ++p) {
            const int m0 = p * 16;
            v8f aR = ZERO8, aZ = ZERO8, aN = ZERO8;
            const __bf16* ap = h1s + (size_t)(m0 + arow) * DH_ + akoff;
            for (int k0 = 0; k0 < DH_; k0 += 32) {
                v16bf a  = ld_afrag(ap + k0);
                v16bf br = ld_bfrag(whh1_p + k0);
                v16bf bz = ld_bfrag(whh1_p + (size_t)512 * DH_ + k0);
                v16bf bn = ld_bfrag(whh1_p + (size_t)1024 * DH_ + k0);
                aR = wmma_bf16(a, br, aR);
                aZ = wmma_bf16(a, bz, aZ);
                aN = wmma_bf16(a, bn, aN);
            }
#pragma unroll
            for (int r = 0; r < 8; ++r) {
                const int b = m0 + mbase + r;
                float gr, gz, gn;
                if (enc) {
                    const float* g = Genc + ((size_t)t * B_ + b) * G3 + jcol;
                    gr = g[0]; gz = g[512]; gn = g[1024];
                } else {
                    gr = bi1_r; gz = bi1_z; gn = bi1_n;
                }
                const float hr = aR[r] + bh1_r;
                const float hz = aZ[r] + bh1_z;
                const float hn = aN[r] + bh1_n;
                const float rg = sigmoidf_(gr + hr);
                const float zg = sigmoidf_(gz + hz);
                const float ng = tanhf(gn + rg * hn);
                const float ho = (float)h1s[(size_t)b * DH_ + jcol];
                nh1[p * 8 + r] = (__bf16)((1.0f - zg) * ng + zg * ho);
            }
        }
        __syncthreads();
#pragma unroll
        for (int p = 0; p < 8; ++p)
#pragma unroll
            for (int r = 0; r < 8; ++r)
                h1s[(size_t)(p * 16 + mbase + r) * DH_ + jcol] = nh1[p * 8 + r];
        __syncthreads();

        // --------- GRU2: gi2a = h1new @ Wih2[:, :512]^T ; gh2 = h2 @ Whh2^T ---------
        __bf16 nh2[64];
#pragma unroll 1
        for (int p = 0; p < 8; ++p) {
            const int m0 = p * 16;
            v8f iR = ZERO8, iZ = ZERO8, iN = ZERO8;
            v8f hR = ZERO8, hZ = ZERO8, hN = ZERO8;
            const __bf16* a1p = h1s + (size_t)(m0 + arow) * DH_ + akoff;
            const __bf16* a2p = h2s + (size_t)(m0 + arow) * DH_ + akoff;
            for (int k0 = 0; k0 < DH_; k0 += 32) {
                v16bf a1 = ld_afrag(a1p + k0);
                v16bf a2 = ld_afrag(a2p + k0);
                v16bf b0 = ld_bfrag(wih2_p + k0);
                iR = wmma_bf16(a1, b0, iR);
                v16bf b1 = ld_bfrag(wih2_p + (size_t)512 * (DH_ + DW_) + k0);
                iZ = wmma_bf16(a1, b1, iZ);
                v16bf b2 = ld_bfrag(wih2_p + (size_t)1024 * (DH_ + DW_) + k0);
                iN = wmma_bf16(a1, b2, iN);
                v16bf b3 = ld_bfrag(whh2_p + k0);
                hR = wmma_bf16(a2, b3, hR);
                v16bf b4 = ld_bfrag(whh2_p + (size_t)512 * DH_ + k0);
                hZ = wmma_bf16(a2, b4, hZ);
                v16bf b5 = ld_bfrag(whh2_p + (size_t)1024 * DH_ + k0);
                hN = wmma_bf16(a2, b5, hN);
            }
#pragma unroll
            for (int r = 0; r < 8; ++r) {
                const int b = m0 + mbase + r;
                float gr, gz, gn;
                if (enc) {
                    gr = bi2_r; gz = bi2_z; gn = bi2_n;
                } else {
                    const float* g = Gword + ((size_t)td * B_ + b) * G3 + jcol;
                    gr = g[0]; gz = g[512]; gn = g[1024];
                }
                const float rg = sigmoidf_(iR[r] + gr + hR[r] + bh2_r);
                const float zg = sigmoidf_(iZ[r] + gz + hZ[r] + bh2_z);
                const float ng = tanhf(iN[r] + gn + rg * (hN[r] + bh2_n));
                const float ho = (float)h2s[(size_t)b * DH_ + jcol];
                const float hv = (1.0f - zg) * ng + zg * ho;
                nh2[p * 8 + r] = (__bf16)hv;
                if (!enc)
                    H2dec[((size_t)b * TDEC + td) * DH_ + jcol] = (__bf16)hv;
            }
        }
        __syncthreads();
#pragma unroll
        for (int p = 0; p < 8; ++p)
#pragma unroll
            for (int r = 0; r < 8; ++r)
                h2s[(size_t)(p * 16 + mbase + r) * DH_ + jcol] = nh2[p * 8 + r];
        __syncthreads();
    }
}

// ---------------------------------------------------------------------------
// In-place log_softmax over rows of 16000 (one block per row)
// ---------------------------------------------------------------------------
__global__ __launch_bounds__(256) void logsoftmax_kernel(float* __restrict__ out) {
    const int row = blockIdx.x;
    float* p = out + (size_t)row * V_;
    __shared__ float red[256];
    const int tid = threadIdx.x;

    float mx = -3.4e38f;
    for (int i = tid; i < V_; i += 256) mx = fmaxf(mx, p[i]);
    red[tid] = mx;
    __syncthreads();
    for (int s = 128; s > 0; s >>= 1) {
        if (tid < s) red[tid] = fmaxf(red[tid], red[tid + s]);
        __syncthreads();
    }
    const float m = red[0];
    __syncthreads();

    float sum = 0.0f;
    for (int i = tid; i < V_; i += 256) sum += __expf(p[i] - m);
    red[tid] = sum;
    __syncthreads();
    for (int s = 128; s > 0; s >>= 1) {
        if (tid < s) red[tid] += red[tid + s];
        __syncthreads();
    }
    const float shift = m + __logf(red[0]);
    __syncthreads();

    for (int i = tid; i < V_; i += 256) p[i] = p[i] - shift;
}

// ---------------------------------------------------------------------------
// Host launcher
// ---------------------------------------------------------------------------
extern "C" void kernel_launch(void* const* d_in, const int* in_sizes, int n_in,
                              void* d_out, int out_size, void* d_ws, size_t ws_size,
                              hipStream_t stream) {
    (void)in_sizes; (void)n_in; (void)out_size; (void)ws_size;
    const float* vid    = (const float*)d_in[0];
    const int*   tgt    = (const int*)  d_in[1];
    const float* emb    = (const float*)d_in[2];
    const float* w_ih1  = (const float*)d_in[3];
    const float* w_hh1  = (const float*)d_in[4];
    const float* b_ih1  = (const float*)d_in[5];
    const float* b_hh1  = (const float*)d_in[6];
    const float* w_ih2  = (const float*)d_in[7];
    const float* w_hh2  = (const float*)d_in[8];
    const float* b_ih2  = (const float*)d_in[9];
    const float* b_hh2  = (const float*)d_in[10];
    const float* out_w  = (const float*)d_in[11];
    const float* out_b  = (const float*)d_in[12];
    float* out = (float*)d_out;

    // element counts
    const size_t N_WIH1 = (size_t)G3 * DV_;          // 3,145,728
    const size_t N_WHH1 = (size_t)G3 * DH_;          //   786,432
    const size_t N_WIH2 = (size_t)G3 * (DH_ + DW_);  // 1,572,864
    const size_t N_WHH2 = (size_t)G3 * DH_;
    const size_t N_OUTW = (size_t)V_ * DH_;          // 8,192,000
    const size_t N_XENC = (size_t)NF_ * B_ * DV_;    // 10,485,760
    const size_t N_WD   = (size_t)TDEC * B_ * DW_;   // 1,769,472
    const size_t N_H2   = (size_t)B_ * TDEC * DH_;   // 1,769,472
    const size_t N_GENC = (size_t)NF_ * B_ * G3;     // 7,864,320 (f32)
    const size_t N_GWRD = (size_t)TDEC * B_ * G3;    // 5,308,416 (f32)

    char* w = (char*)d_ws;
    size_t off = 0;
    auto take = [&](size_t bytes) -> void* {
        void* p = w + off;
        off += (bytes + 255) & ~(size_t)255;
        return p;
    };
    __bf16* wih1b = (__bf16*)take(N_WIH1 * 2);
    __bf16* whh1b = (__bf16*)take(N_WHH1 * 2);
    __bf16* wih2b = (__bf16*)take(N_WIH2 * 2);
    __bf16* whh2b = (__bf16*)take(N_WHH2 * 2);
    __bf16* outwb = (__bf16*)take(N_OUTW * 2);
    __bf16* xencb = (__bf16*)take(N_XENC * 2);
    __bf16* wdb   = (__bf16*)take(N_WD * 2);
    __bf16* h2b   = (__bf16*)take(N_H2 * 2);
    float*  genc  = (float*) take(N_GENC * 4);
    float*  gword = (float*) take(N_GWRD * 4);

    // 1) weight conversions fp32 -> bf16
    f2bf_kernel<<<(int)((N_WIH1 + 255) / 256), 256, 0, stream>>>(w_ih1, wih1b, (int)N_WIH1);
    f2bf_kernel<<<(int)((N_WHH1 + 255) / 256), 256, 0, stream>>>(w_hh1, whh1b, (int)N_WHH1);
    f2bf_kernel<<<(int)((N_WIH2 + 255) / 256), 256, 0, stream>>>(w_ih2, wih2b, (int)N_WIH2);
    f2bf_kernel<<<(int)((N_WHH2 + 255) / 256), 256, 0, stream>>>(w_hh2, whh2b, (int)N_WHH2);
    f2bf_kernel<<<(int)((N_OUTW + 255) / 256), 256, 0, stream>>>(out_w, outwb, (int)N_OUTW);

    // 2) activation prep
    permute_vid_kernel<<<(int)((N_XENC + 255) / 256), 256, 0, stream>>>(vid, xencb);
    gather_words_kernel<<<(int)((N_WD + 255) / 256), 256, 0, stream>>>(tgt, emb, wdb);

    // 3) Genc = Xenc @ w_ih1^T + b_ih1   [5120,1536,K=2048]
    gemm_bf16_wmma<<<dim3(G3 / 128, (NF_ * B_) / 64), 256, 0, stream>>>(
        xencb, DV_, wih1b, DV_, b_ih1, genc, G3, DV_);

    // 4) Gword = words @ w_ih2[:,512:]^T + b_ih2   [3456,1536,K=512]
    gemm_bf16_wmma<<<dim3(G3 / 128, (TDEC * B_) / 64), 256, 0, stream>>>(
        wdb, DW_, wih2b + DH_, DH_ + DW_, b_ih2, gword, G3, DW_);

    // 5) recurrent scan (single WGP-resident workgroup, 32 waves)
    recurrent_kernel<<<1, 1024, 0, stream>>>(
        whh1b, wih2b, whh2b, genc, gword,
        b_ih1, b_hh1, b_ih2, b_hh2, h2b);

    // 6) logits = H2dec @ out_w^T + out_b   [3456,16000,K=512] -> d_out
    gemm_bf16_wmma<<<dim3(V_ / 128, (B_ * TDEC) / 64), 256, 0, stream>>>(
        h2b, DH_, outwb, DH_, out_b, out, V_, DH_);

    // 7) in-place log_softmax per row
    logsoftmax_kernel<<<B_ * TDEC, 256, 0, stream>>>(out);
}
